// Lstm_88064009437726
// MI455X (gfx1250) — matrix operand order repaired and checked by reading, not compiled, not run on hardware
//
#include <hip/hip_runtime.h>
#include <math.h>

// ---------------------------------------------------------------------------
// Problem dimensions (fixed by the reference)
// ---------------------------------------------------------------------------
#define T_DIM 200
#define B_DIM 64
#define V_DIM 256
#define E_DIM 512
#define NN_DIM 1024
#define ON_DIM 1024

typedef __attribute__((ext_vector_type(16))) __bf16 v16bf;
typedef __attribute__((ext_vector_type(8)))  float  v8f;

// ---------------------------------------------------------------------------
// WMMA helpers (CDNA5 wave32, v_wmma_f32_16x16x32_bf16)
// ---------------------------------------------------------------------------
__device__ __forceinline__ v8f wmma_bf16(v16bf a, v16bf b, v8f c) {
    // (neg_a, A, neg_b, B, c_mod, C, reuse_a, reuse_b)
    return __builtin_amdgcn_wmma_f32_16x16x32_bf16(false, a, false, b,
                                                   (short)0, c, false, false);
}

// A-fragment: 16x32 bf16, built on the fly from row-major fp32.
// ISA layout: lanes 0-15 (M=lane): K = e (e<8), e+8 (e>=8)
//             lanes 16-31 (M=lane-16): K = e+8 (e<8), e+16 (e>=8)
// => per lane: two contiguous 8-float runs at k-offsets {0,16} or {8,24}.
__device__ __forceinline__ v16bf load_a_frag(const float* __restrict__ A,
                                             int lda, int row, int kb, int lane) {
    const float* p = A + (size_t)row * lda + kb + ((lane & 16) ? 8 : 0);
    float4 a0 = *(const float4*)(p);
    float4 a1 = *(const float4*)(p + 4);
    float4 a2 = *(const float4*)(p + 16);
    float4 a3 = *(const float4*)(p + 20);
    v16bf r;
    r[0]  = (__bf16)a0.x; r[1]  = (__bf16)a0.y; r[2]  = (__bf16)a0.z; r[3]  = (__bf16)a0.w;
    r[4]  = (__bf16)a1.x; r[5]  = (__bf16)a1.y; r[6]  = (__bf16)a1.z; r[7]  = (__bf16)a1.w;
    r[8]  = (__bf16)a2.x; r[9]  = (__bf16)a2.y; r[10] = (__bf16)a2.z; r[11] = (__bf16)a2.w;
    r[12] = (__bf16)a3.x; r[13] = (__bf16)a3.y; r[14] = (__bf16)a3.z; r[15] = (__bf16)a3.w;
    return r;
}

// B-fragment: pre-packed, 512 bf16 per (32x16) tile, 32 B contiguous per lane.
__device__ __forceinline__ v16bf load_b_frag(const __bf16* __restrict__ Bf,
                                             size_t fragIdx, int lane) {
    return *(const v16bf*)(Bf + fragIdx * 512 + (size_t)lane * 16);
}

// Async copy 16B/lane from global to LDS, tracked by ASYNCcnt.
// VDST = per-lane LDS byte address, VADDR = per-lane 64-bit global address;
// INST_OFFSET is added on both sides (ISA 15.18.3).
__device__ __forceinline__ void async_g2l_b128(unsigned int lds_addr, const void* gaddr) {
    asm volatile("global_load_async_to_lds_b128 %0, %1, off"
                 :: "v"(lds_addr), "v"(gaddr) : "memory");
}
__device__ __forceinline__ void async_g2l_b128_off16(unsigned int lds_addr, const void* gaddr) {
    asm volatile("global_load_async_to_lds_b128 %0, %1, off offset:16"
                 :: "v"(lds_addr), "v"(gaddr) : "memory");
}
__device__ __forceinline__ void wait_async0() {
    asm volatile("s_wait_asynccnt 0x0" ::: "memory");
}

// ---------------------------------------------------------------------------
// Pack a row-major fp32 weight (K x N) into bf16 WMMA B-fragments.
// Fragment order: frag = nt * KT + kt  (n-major so the GEMM k-loop streams).
// Within a fragment: lane = idx/16, e = idx%16;
//   n = nt*16 + (lane&15);  k = kt*32 + ((lane&16)?8:0) + (e<8 ? e : e+8)
// ---------------------------------------------------------------------------
__global__ void pack_b_kernel(const float* __restrict__ W,
                              __bf16* __restrict__ dst, int K, int N) {
    int KT = K >> 5;
    long idx = (long)blockIdx.x * blockDim.x + threadIdx.x;
    long total = (long)K * N;
    if (idx >= total) return;
    int f = (int)(idx >> 9);
    int r = (int)(idx & 511);
    int lane = r >> 4, e = r & 15;
    int nt = f / KT, kt = f % KT;
    int n = nt * 16 + (lane & 15);
    int k = kt * 32 + ((lane & 16) ? 8 : 0) + (e < 8 ? e : e + 8);
    dst[idx] = (__bf16)W[(size_t)k * N + n];
}

// ---------------------------------------------------------------------------
// Copy initial LSTM state into workspace ping-pong buffers.
// ---------------------------------------------------------------------------
__global__ void init_state_kernel(const float* __restrict__ h0, const float* __restrict__ c0,
                                  const float* __restrict__ h1, const float* __restrict__ c1,
                                  float* __restrict__ H0, float* __restrict__ C0,
                                  float* __restrict__ H1, float* __restrict__ C1, int n) {
    int i = blockIdx.x * blockDim.x + threadIdx.x;
    if (i < n) { H0[i] = h0[i]; C0[i] = c0[i]; H1[i] = h1[i]; C1[i] = c1[i]; }
}

// ---------------------------------------------------------------------------
// Generic WMMA GEMM:  Cout(MxN) = A(MxK, fp32 row-major) * Bpacked + bias
// Wave tile = 16 rows x 64 cols (4 n-tiles). 8 waves / block.
// Used for: embedding GEMM, head GEMM 0 (+ReLU), head GEMM 1.
// ---------------------------------------------------------------------------
__global__ __launch_bounds__(256) void gemm_kernel(
    const float* __restrict__ A, const __bf16* __restrict__ Bf,
    const float* __restrict__ bias, float* __restrict__ Cout,
    int M, int N, int K, int do_relu) {
    int lane = threadIdx.x & 31;
    int wid  = blockIdx.x * 8 + (threadIdx.x >> 5);
    int numNB = N >> 6;
    int total = (M >> 4) * numNB;
    if (wid >= total) return;                 // wave-uniform
    int mt = wid / numNB, nb = wid % numNB;
    int KT = K >> 5;
    int mrow = mt * 16 + (lane & 15);

    v8f acc[4] = {};
    for (int kt = 0; kt < KT; ++kt) {
        v16bf a = load_a_frag(A, K, mrow, kt * 32, lane);
        if (kt + 1 < KT)  // warm GL2 for the next k-slab of B fragments
            __builtin_prefetch(Bf + ((size_t)(nb * 4) * KT + kt + 1) * 512, 0, 0);
#pragma unroll
        for (int j = 0; j < 4; ++j) {
            size_t ntg = (size_t)(nb * 4 + j);
            v16bf b = load_b_frag(Bf, ntg * KT + kt, lane);
            acc[j] = wmma_bf16(a, b, acc[j]);
        }
    }
#pragma unroll
    for (int j = 0; j < 4; ++j) {
        int col = nb * 64 + j * 16 + (lane & 15);
        float bv = bias ? bias[col] : 0.0f;
#pragma unroll
        for (int r = 0; r < 8; ++r) {
            int row = mt * 16 + r + ((lane & 16) ? 8 : 0);
            float v = acc[j][r] + bv;
            if (do_relu) v = fmaxf(v, 0.0f);
            Cout[(size_t)row * N + col] = v;
        }
    }
}

// ---------------------------------------------------------------------------
// Fused LSTM cell:  S = [X | Hin] @ W + b ; gates ; update C ; emit Hout.
// Grid: 16 blocks x 256 threads (8 waves). Block `wg` owns hidden columns
// [wg*64, wg*64+64) and computes that slice of all FOUR gates so the
// nonlinearity fuses after one barrier.
//
// Weight streaming: double-buffered async copies (GLOBAL_LOAD_ASYNC_TO_LDS,
// ASYNCcnt) of the 16 (gate, n-tile) fragment rows, CHUNK=4 k-tiles at a
// time. Chunk c+1 streams from L2 while the WMMAs consume chunk c from LDS.
// LDS: 2 x 64KB staging buffers, reused afterwards as the 4x64x65 fp32 gate
// buffer. Dynamic shared = 128KB (< 320KB WGP budget).
// ---------------------------------------------------------------------------
#define CELL_CHUNK 4
#define CELL_BUFB  (16 * CELL_CHUNK * 1024)      // 64KB per staging buffer
#define CELL_LDS_BYTES (2 * CELL_BUFB)           // 128KB dynamic LDS

__global__ __launch_bounds__(256) void lstm_cell_kernel(
    const float* __restrict__ X, int ldx, int K0,
    const float* __restrict__ Hin,
    float* __restrict__ Hout,
    float* __restrict__ C,
    const float* __restrict__ bias,
    const __bf16* __restrict__ Wf,
    float* __restrict__ rnn_out /* nullable; pre-offset by t*B*NN */) {
    extern __shared__ char smem[];

    int lane = threadIdx.x & 31;
    int wave = threadIdx.x >> 5;
    int g  = wave >> 1;               // gate 0..3 (consumer role)
    int mh = wave & 1;                // m-half: rows [mh*32, mh*32+32)
    int wg = blockIdx.x;              // 0..15 -> hidden column block
    int n0 = wg * 64;
    int KT  = (K0 + NN_DIM) >> 5;
    int KT0 = K0 >> 5;
    int NC  = KT / CELL_CHUNK;

    // Low 32 bits of the flat shared address == workgroup-relative LDS offset.
    unsigned int lds_base = (unsigned int)(uintptr_t)&smem[0];

    // Each wave async-loads fragment rows {2*wave, 2*wave+1} of 16.
    auto issue_chunk = [&](int c, int buf) {
#pragma unroll
        for (int rr = 0; rr < 2; ++rr) {
            int r  = wave * 2 + rr;
            int gg = r >> 2, j = r & 3;
            size_t ntg = (size_t)(gg * 64 + wg * 4 + j);   // of 256 n-tiles
#pragma unroll
            for (int ktl = 0; ktl < CELL_CHUNK; ++ktl) {
                int kt = c * CELL_CHUNK + ktl;
                const char* src =
                    (const char*)(Wf + (ntg * (size_t)KT + kt) * 512) + lane * 32;
                unsigned int dst = lds_base + (unsigned int)(buf * CELL_BUFB +
                                   (r * CELL_CHUNK + ktl) * 1024 + lane * 32);
                async_g2l_b128(dst, src);
                async_g2l_b128_off16(dst, src);
            }
        }
    };

    v8f acc[2][4] = {};
    issue_chunk(0, 0);
    for (int c = 0; c < NC; ++c) {
        wait_async0();                 // my chunk-c copies landed
        __syncthreads();               // everyone's chunk-c copies landed
        if (c + 1 < NC) issue_chunk(c + 1, (c + 1) & 1);
        int buf = c & 1;
#pragma unroll
        for (int ktl = 0; ktl < CELL_CHUNK; ++ktl) {
            int kt = c * CELL_CHUNK + ktl;
            const float* Ap; int lda, kb;
            if (kt < KT0) { Ap = X;   lda = ldx;    kb = kt * 32; }
            else          { Ap = Hin; lda = NN_DIM; kb = (kt - KT0) * 32; }
            v16bf a0 = load_a_frag(Ap, lda, mh * 32      + (lane & 15), kb, lane);
            v16bf a1 = load_a_frag(Ap, lda, mh * 32 + 16 + (lane & 15), kb, lane);
#pragma unroll
            for (int j = 0; j < 4; ++j) {
                int r = g * 4 + j;
                const v16bf* bp = (const v16bf*)(smem + buf * CELL_BUFB +
                                  (r * CELL_CHUNK + ktl) * 1024 + lane * 32);
                v16bf b = *bp;
                acc[0][j] = wmma_bf16(a0, b, acc[0][j]);
                acc[1][j] = wmma_bf16(a1, b, acc[1][j]);
            }
        }
        __syncthreads();               // done reading buf before chunk c+2 overwrites
    }

    // Reuse LDS as gate-preactivation buffer sS[4][64][65] (66,560 B).
    float* sS = (float*)smem;
    // D layout: VGPR r -> row r + 8*(lane>=16), col = lane&15
#pragma unroll
    for (int i = 0; i < 2; ++i)
#pragma unroll
        for (int j = 0; j < 4; ++j)
#pragma unroll
            for (int r = 0; r < 8; ++r) {
                int row = mh * 32 + i * 16 + r + ((lane & 16) ? 8 : 0);
                int col = j * 16 + (lane & 15);
                sS[(g * 64 + row) * 65 + col] = acc[i][j][r];
            }
    __syncthreads();

    // Gate math: s[:, :3NN]->sigmoid (f,i,o); s[:, 3NN:]->tanh
    for (int e = threadIdx.x; e < 64 * 64; e += 256) {
        int b  = e >> 6;
        int nl = e & 63;
        int n  = n0 + nl;
        float fg = sS[(0 * 64 + b) * 65 + nl] + bias[n];
        float ig = sS[(1 * 64 + b) * 65 + nl] + bias[NN_DIM + n];
        float og = sS[(2 * 64 + b) * 65 + nl] + bias[2 * NN_DIM + n];
        float tg = sS[(3 * 64 + b) * 65 + nl] + bias[3 * NN_DIM + n];
        fg = 1.0f / (1.0f + __expf(-fg));
        ig = 1.0f / (1.0f + __expf(-ig));
        og = 1.0f / (1.0f + __expf(-og));
        float cv = fg * C[(size_t)b * NN_DIM + n] + ig * tanhf(tg);
        C[(size_t)b * NN_DIM + n] = cv;
        float hv = og * tanhf(cv);
        Hout[(size_t)b * NN_DIM + n] = hv;
        if (rnn_out) rnn_out[(size_t)b * NN_DIM + n] = hv;
    }
}

// ---------------------------------------------------------------------------
// Host launch
// ---------------------------------------------------------------------------
static inline size_t align256(size_t x) { return (x + 255) & ~(size_t)255; }

extern "C" void kernel_launch(void* const* d_in, const int* in_sizes, int n_in,
                              void* d_out, int out_size, void* d_ws, size_t ws_size,
                              hipStream_t stream) {
    (void)in_sizes; (void)n_in; (void)out_size; (void)ws_size;
    const float* inputs  = (const float*)d_in[0];   // T,B,V
    const float* h0_in   = (const float*)d_in[1];
    const float* c0_in   = (const float*)d_in[2];
    const float* h1_in   = (const float*)d_in[3];
    const float* c1_in   = (const float*)d_in[4];
    const float* embW    = (const float*)d_in[5];   // V,E
    const float* w0      = (const float*)d_in[6];   // E+NN, 4NN
    const float* b0      = (const float*)d_in[7];
    const float* w1      = (const float*)d_in[8];   // 2NN, 4NN
    const float* b1      = (const float*)d_in[9];
    const float* ow0     = (const float*)d_in[10];  // NN, ON
    const float* ob0     = (const float*)d_in[11];
    const float* ow1     = (const float*)d_in[12];  // ON, V
    const float* ob1     = (const float*)d_in[13];
    float* logits = (float*)d_out;                  // (T*B, V)

    char* ws = (char*)d_ws;
    size_t off = 0;
    auto carve = [&](size_t bytes) { size_t o = off; off = align256(off + bytes); return o; };

    const size_t nEmbW = (size_t)V_DIM * E_DIM;
    const size_t nW0   = (size_t)(E_DIM + NN_DIM) * 4 * NN_DIM;
    const size_t nW1   = (size_t)(2 * NN_DIM) * 4 * NN_DIM;
    const size_t nOW0  = (size_t)NN_DIM * ON_DIM;
    const size_t nOW1  = (size_t)ON_DIM * V_DIM;
    const size_t nState = (size_t)B_DIM * NN_DIM;
    const size_t nEmbAct = (size_t)T_DIM * B_DIM * E_DIM;
    const size_t nRnn    = (size_t)T_DIM * B_DIM * NN_DIM;
    const size_t nHid    = (size_t)T_DIM * B_DIM * ON_DIM;

    __bf16* embW_f = (__bf16*)(ws + carve(nEmbW * 2));
    __bf16* w0_f   = (__bf16*)(ws + carve(nW0 * 2));
    __bf16* w1_f   = (__bf16*)(ws + carve(nW1 * 2));
    __bf16* ow0_f  = (__bf16*)(ws + carve(nOW0 * 2));
    __bf16* ow1_f  = (__bf16*)(ws + carve(nOW1 * 2));
    float* emb_act = (float*)(ws + carve(nEmbAct * 4));
    float* H0buf   = (float*)(ws + carve(2 * nState * 4));  // ping-pong
    float* C0buf   = (float*)(ws + carve(nState * 4));
    float* H1buf   = (float*)(ws + carve(2 * nState * 4));  // ping-pong
    float* C1buf   = (float*)(ws + carve(nState * 4));
    float* rnn_out = (float*)(ws + carve(nRnn * 4));
    float* hidden  = (float*)(ws + carve(nHid * 4));

    // 1) pack all weights into WMMA B-fragment layout (bf16)
    {
        int thr = 256;
        pack_b_kernel<<<(int)((nEmbW + thr - 1) / thr), thr, 0, stream>>>(embW, embW_f, V_DIM, E_DIM);
        pack_b_kernel<<<(int)((nW0   + thr - 1) / thr), thr, 0, stream>>>(w0,   w0_f,   E_DIM + NN_DIM, 4 * NN_DIM);
        pack_b_kernel<<<(int)((nW1   + thr - 1) / thr), thr, 0, stream>>>(w1,   w1_f,   2 * NN_DIM,     4 * NN_DIM);
        pack_b_kernel<<<(int)((nOW0  + thr - 1) / thr), thr, 0, stream>>>(ow0,  ow0_f,  NN_DIM, ON_DIM);
        pack_b_kernel<<<(int)((nOW1  + thr - 1) / thr), thr, 0, stream>>>(ow1,  ow1_f,  ON_DIM, V_DIM);
    }

    // 2) initial state
    init_state_kernel<<<(int)((nState + 255) / 256), 256, 0, stream>>>(
        h0_in, c0_in, h1_in, c1_in, H0buf, C0buf, H1buf, C1buf, (int)nState);

    // 3) embedding GEMM: (T*B x V) * (V x E) -> emb_act
    {
        int M = T_DIM * B_DIM, N = E_DIM, K = V_DIM;
        int waves = (M >> 4) * (N >> 6);
        gemm_kernel<<<(waves + 7) / 8, 256, 0, stream>>>(
            inputs, embW_f, nullptr, emb_act, M, N, K, 0);
    }

    // 4) sequential scan: 2 fused cell kernels per timestep
    for (int t = 0; t < T_DIM; ++t) {
        int rp = t & 1, wp = rp ^ 1;
        // cell 0: x = emb[t] (B x E), h = H0buf[rp] -> H0buf[wp]
        lstm_cell_kernel<<<16, 256, CELL_LDS_BYTES, stream>>>(
            emb_act + (size_t)t * B_DIM * E_DIM, E_DIM, E_DIM,
            H0buf + (size_t)rp * nState,
            H0buf + (size_t)wp * nState,
            C0buf, b0, w0_f, nullptr);
        // cell 1: x = h0_new (B x NN), h = H1buf[rp] -> H1buf[wp]; emit rnn_out[t]
        lstm_cell_kernel<<<16, 256, CELL_LDS_BYTES, stream>>>(
            H0buf + (size_t)wp * nState, NN_DIM, NN_DIM,
            H1buf + (size_t)rp * nState,
            H1buf + (size_t)wp * nState,
            C1buf, b1, w1_f,
            rnn_out + (size_t)t * B_DIM * NN_DIM);
    }

    // 5) head GEMM 0: relu(rnn_out @ out_w0 + out_b0) -> hidden
    {
        int M = T_DIM * B_DIM, N = ON_DIM, K = NN_DIM;
        int waves = (M >> 4) * (N >> 6);
        gemm_kernel<<<(waves + 7) / 8, 256, 0, stream>>>(
            rnn_out, ow0_f, ob0, hidden, M, N, K, 1);
    }
    // 6) head GEMM 1: hidden @ out_w1 + out_b1 -> logits
    {
        int M = T_DIM * B_DIM, N = V_DIM, K = ON_DIM;
        int waves = (M >> 4) * (N >> 6);
        gemm_kernel<<<(waves + 7) / 8, 256, 0, stream>>>(
            hidden, ow1_f, ob1, logits, M, N, K, 0);
    }
}
